// lowrank_RNN_72945724555288
// MI455X (gfx1250) — matrix operand order repaired and checked by reading, not compiled, not run on hardware
//
#include <hip/hip_runtime.h>
#include <math.h>

#define NN 1024
#define RR 6
#define BB 64
#define TT 400

typedef __bf16          v16bf  __attribute__((ext_vector_type(16)));
typedef float           v8f    __attribute__((ext_vector_type(8)));
typedef unsigned short  u16x8  __attribute__((ext_vector_type(8)));
typedef unsigned short  u16x16 __attribute__((ext_vector_type(16)));

union BF16Frag { v16bf bf; u16x16 u; u16x8 h[2]; };

// float -> bf16 with round-to-nearest-even (stored as raw u16)
__device__ __forceinline__ unsigned short f2bf(float f) {
  unsigned int u = __float_as_uint(f);
  unsigned int r = u + 0x7FFFu + ((u >> 16) & 1u);
  return (unsigned short)(r >> 16);
}

// ---------------------------------------------------------------------------
// Kernel 1: Wb[i*N+j] = bf16( noise[i][j] + (1/N) * sum_r rV[r][i]*lV[r][j] )
// ---------------------------------------------------------------------------
__global__ void wb_kernel(const float* __restrict__ lV, const float* __restrict__ rV,
                          const float* __restrict__ noiseM, unsigned short* __restrict__ Wb) {
  int tid = blockIdx.x * blockDim.x + threadIdx.x;   // 0 .. N*N-1
  int i = tid >> 10;        // row (output neuron)
  int j = tid & (NN - 1);   // col (input neuron)
  float s = noiseM[tid];
#pragma unroll
  for (int r = 0; r < RR; ++r)
    s = fmaf(rV[r * NN + i] * (1.0f / NN), lV[r * NN + j], s);
  Wb[tid] = f2bf(s);
}

// ---------------------------------------------------------------------------
// Kernel 2: h[b][n] = hidden0[n];  A0[b][n] = bf16(tanh(hidden0[n]))
// ---------------------------------------------------------------------------
__global__ void init_kernel(const float* __restrict__ hidden0,
                            float* __restrict__ h, unsigned short* __restrict__ A0) {
  int tid = blockIdx.x * blockDim.x + threadIdx.x;   // 0 .. B*N-1
  int n = tid & (NN - 1);
  float hv = hidden0[n];
  h[tid] = hv;
  A0[tid] = f2bf(tanhf(hv));
}

// ---------------------------------------------------------------------------
// Kernel 3: one recurrence step. One wave32 per 16x16 output tile.
//   mm = Aread @ Wb^T ;  hNew = 0.9 h + 0.1 (mm + x_t)
//   hidden_out[t] = tanh(hNew);  Awrite = bf16(tanh(hNew))
// Fragments are built straight from global memory in the ISA VGPR layout:
//   A 16x32 bf16 : lane m (0-15) k = k0+0..7 & k0+16..23 ; lanes 16-31 +8
//   B 32x16 bf16 : lane j (0-15) k = k0+0..7 & k0+16..23 ; lanes 16-31 +8
//   C 16x16 f32  : VGPR v holds M = v (lanes 0-15) / v+8 (lanes 16-31), N = lane&15
// ---------------------------------------------------------------------------
__global__ void __launch_bounds__(32) step_kernel(
    const unsigned short* __restrict__ Aread,
    unsigned short* __restrict__ Awrite,
    const unsigned short* __restrict__ Wb,
    float* __restrict__ h,
    const float* __restrict__ Input,      // [B][T][N]
    float* __restrict__ hidden_out,       // [T][B][N]
    int t) {
  const int lane  = threadIdx.x & 31;
  const int btile = blockIdx.x & 3;    // 4 batch tiles of 16
  const int ntile = blockIdx.x >> 2;   // 64 output-neuron tiles of 16
  const bool hi   = lane >= 16;
  const int  l16  = lane & 15;

  const unsigned short* aBase = Aread + (size_t)(btile * 16 + l16) * NN + (hi ? 8 : 0);
  const unsigned short* bBase = Wb    + (size_t)(ntile * 16 + l16) * NN + (hi ? 8 : 0);

  v8f c = {};
  for (int k0 = 0; k0 < NN; k0 += 32) {
    BF16Frag a, b;
    a.h[0] = *(const u16x8*)(aBase + k0);        // k0 + (hi?8:0) + 0..7
    a.h[1] = *(const u16x8*)(aBase + k0 + 16);   // k0 + (hi?8:0) + 16..23
    b.h[0] = *(const u16x8*)(bBase + k0);        // k0 + (hi?8:0) + 0..7
    b.h[1] = *(const u16x8*)(bBase + k0 + 16);   // k0 + (hi?8:0) + 16..23
    if (k0 + 32 < NN) {                          // pull next W line toward WGP$
      __builtin_prefetch(bBase + k0 + 32, 0, 1);
    }
    c = __builtin_amdgcn_wmma_f32_16x16x32_bf16(
            /*neg_a=*/false, a.bf, /*neg_b=*/false, b.bf,
            /*c_mod=*/(short)0, c, /*reuse_a=*/false, /*reuse_b=*/false);
  }

  const int ncol = ntile * 16 + l16;
#pragma unroll
  for (int v = 0; v < 8; ++v) {
    int brow = btile * 16 + v + (hi ? 8 : 0);
    size_t idx = (size_t)brow * NN + ncol;
    float hOld = h[idx];
    float x    = Input[((size_t)brow * TT + t) * NN + ncol];
    float hNew = 0.9f * hOld + 0.1f * (c[v] + x);
    float aNew = tanhf(hNew);
    h[idx] = hNew;
    Awrite[idx] = f2bf(aNew);
    hidden_out[(size_t)t * BB * NN + idx] = aNew;
  }
}

// ---------------------------------------------------------------------------
// Kernel 4: geometry = tanh(hidden) @ gW^T + gb ; readout = geometry @ rW^T
// One wave32 per (t,b); strided loads + __shfl_xor wave reduction.
// ---------------------------------------------------------------------------
__global__ void head_kernel(const float* __restrict__ hidden,   // [T*B][N]
                            const float* __restrict__ gW,       // [2][N]
                            const float* __restrict__ gb,       // [2]
                            const float* __restrict__ rW,       // [6][2]
                            float* __restrict__ gOut,           // [T*B][2]
                            float* __restrict__ rOut) {         // [T*B][6]
  int gw   = (blockIdx.x * blockDim.x + threadIdx.x) >> 5;
  int lane = threadIdx.x & 31;
  if (gw >= TT * BB) return;
  const float* hrow = hidden + (size_t)gw * NN;
  float s0 = 0.0f, s1 = 0.0f;
  for (int k = lane; k < NN; k += 32) {
    float v = tanhf(hrow[k]);
    s0 = fmaf(v, gW[k], s0);
    s1 = fmaf(v, gW[NN + k], s1);
  }
#pragma unroll
  for (int off = 16; off > 0; off >>= 1) {
    s0 += __shfl_xor(s0, off, 32);
    s1 += __shfl_xor(s1, off, 32);
  }
  if (lane == 0) {
    float g0 = s0 + gb[0];
    float g1 = s1 + gb[1];
    gOut[gw * 2 + 0] = g0;
    gOut[gw * 2 + 1] = g1;
#pragma unroll
    for (int c6 = 0; c6 < 6; ++c6)
      rOut[gw * 6 + c6] = g0 * rW[c6 * 2 + 0] + g1 * rW[c6 * 2 + 1];
  }
}

// ---------------------------------------------------------------------------
extern "C" void kernel_launch(void* const* d_in, const int* in_sizes, int n_in,
                              void* d_out, int out_size, void* d_ws, size_t ws_size,
                              hipStream_t stream) {
  (void)in_sizes; (void)n_in; (void)out_size; (void)ws_size;

  const float* hidden0 = (const float*)d_in[0];   // [N]
  const float* Input   = (const float*)d_in[1];   // [B,T,N]
  const float* lV      = (const float*)d_in[2];   // [R,N]
  const float* rV      = (const float*)d_in[3];   // [R,N]
  const float* noiseM  = (const float*)d_in[4];   // [N,N]
  const float* gW      = (const float*)d_in[5];   // [2,N]
  const float* gb      = (const float*)d_in[6];   // [2]
  const float* rW      = (const float*)d_in[7];   // [6,2]

  float* out        = (float*)d_out;
  float* hidden_out = out;                               // [T,B,N]
  float* gOut       = out + (size_t)TT * BB * NN;        // [T,B,2]
  float* rOut       = gOut + (size_t)TT * BB * 2;        // [T,B,6]

  // Workspace layout (2.5 MB total):
  //   [0, 2MB)                : Wb   bf16 [N][N]
  //   [2MB, 2MB+256KB)        : h    f32  [B][N]
  //   then two bf16 A buffers : A0,A1 [B][N] (128 KB each)
  char* ws = (char*)d_ws;
  unsigned short* Wb = (unsigned short*)ws;
  float* h           = (float*)(ws + (size_t)NN * NN * 2);
  unsigned short* A0 = (unsigned short*)(ws + (size_t)NN * NN * 2 + (size_t)BB * NN * 4);
  unsigned short* A1 = A0 + (size_t)BB * NN;

  wb_kernel<<<(NN * NN) / 256, 256, 0, stream>>>(lV, rV, noiseM, Wb);
  init_kernel<<<(BB * NN) / 256, 256, 0, stream>>>(hidden0, h, A0);

  unsigned short* bufs[2] = {A0, A1};
  for (int t = 0; t < TT; ++t) {
    step_kernel<<<256, 32, 0, stream>>>(bufs[t & 1], bufs[(t + 1) & 1],
                                        Wb, h, Input, hidden_out, t);
  }

  head_kernel<<<(TT * BB) / 8, 256, 0, stream>>>(hidden_out, gW, gb, rW, gOut, rOut);
}
